// MotionNet3D_11450382811204
// MI455X (gfx1250) — matrix operand order repaired and checked by reading, not compiled, not run on hardware
//
#include <hip/hip_runtime.h>
#include <hip/hip_bf16.h>

// ---------------------------------------------------------------------------
// MotionNet3D (FlowNet3D-style) forward for gfx1250.
// All 1x1-conv MLP layers run through one generic WMMA GEMM kernel:
//   Y = relu(((X*W^T + b) - mean) * gamma*rsqrt(var+eps) + beta)
// using v_wmma_f32_16x16x32_f16 with f32->f16 staged tiles in LDS.
// Main k-loop is branch/clamp-free (pointer increments only); the ragged
// cin tail is handled by a single clamped k-step.
// ---------------------------------------------------------------------------

typedef __attribute__((ext_vector_type(16))) _Float16 v16h;
typedef __attribute__((ext_vector_type(8)))  _Float16 v8h;
typedef __attribute__((ext_vector_type(8)))  float    v8f;

#define NB 8
#define NP 4096
#define BN_EPS 1e-5f

// --------------------------------- kernels ---------------------------------

// [B,C,N] -> [B,N,C]
__global__ void chlast_ker(const float* __restrict__ in, float* __restrict__ out,
                           int C, int N, long total) {
  long i = (long)blockIdx.x * blockDim.x + threadIdx.x;
  if (i >= total) return;
  int c = (int)(i % C);
  long t = i / C;
  int nn = (int)(t % N);
  int b  = (int)(t / N);
  out[i] = in[((size_t)b * C + c) * N + nn];
}

// Farthest point sampling. One block per batch.
__global__ void fps_ker(const float* __restrict__ xyz, int n, int npoint,
                        int* __restrict__ outidx, float* __restrict__ distws) {
  int b = blockIdx.x;
  const float* p = xyz + (size_t)b * n * 3;
  float* dist = distws + (size_t)b * n;
  int* oi = outidx + (size_t)b * npoint;
  __shared__ float rv[256];
  __shared__ int   ri[256];
  __shared__ int   sfar;
  int tid = threadIdx.x;
  if (tid == 0) sfar = 0;
  for (int i = tid; i < n; i += 256) dist[i] = 1e10f;
  __syncthreads();
  for (int j = 0; j < npoint; ++j) {
    int far = sfar;
    if (tid == 0) oi[j] = far;
    float cx = p[far * 3 + 0], cy = p[far * 3 + 1], cz = p[far * 3 + 2];
    float bv = -1.0f; int bi = n - 1;
    for (int i = tid; i < n; i += 256) {
      float dx = p[i * 3 + 0] - cx;
      float dy = p[i * 3 + 1] - cy;
      float dz = p[i * 3 + 2] - cz;
      float d2 = dx * dx + dy * dy + dz * dz;
      float dm = fminf(dist[i], d2);
      dist[i] = dm;
      if (dm > bv || (dm == bv && i < bi)) { bv = dm; bi = i; }
    }
    rv[tid] = bv; ri[tid] = bi;
    __syncthreads();
    for (int s = 128; s > 0; s >>= 1) {
      if (tid < s) {
        if (rv[tid + s] > rv[tid] ||
            (rv[tid + s] == rv[tid] && ri[tid + s] < ri[tid])) {
          rv[tid] = rv[tid + s]; ri[tid] = ri[tid + s];
        }
      }
      __syncthreads();
    }
    if (tid == 0) sfar = ri[0];
    __syncthreads();
  }
}

// new_xyz[b,s,c] = xyz[b, idx[b,s], c]
__global__ void gather_xyz_ker(const float* __restrict__ xyz, const int* __restrict__ idx,
                               int n, int S, float* __restrict__ out, long total) {
  long i = (long)blockIdx.x * blockDim.x + threadIdx.x;
  if (i >= total) return;
  int c = (int)(i % 3);
  long t = i / 3;
  int s = (int)(t % S);
  int b = (int)(t / S);
  int id = idx[(size_t)b * S + s];
  if (id >= n) id = n - 1;
  out[i] = xyz[((size_t)b * n + id) * 3 + c];
}

// query_ball_point: first <=K indices (ascending) with d2 <= r2, pad with first.
__global__ void ballq_ker(const float* __restrict__ xyz, const float* __restrict__ nxyz,
                          int n, int S, int K, float r2, int* __restrict__ out, long total) {
  long i = (long)blockIdx.x * blockDim.x + threadIdx.x;
  if (i >= total) return;
  int s = (int)(i % S);
  int b = (int)(i / S);
  const float* P = xyz + (size_t)b * n * 3;
  const float* q = nxyz + ((size_t)b * S + s) * 3;
  float qx = q[0], qy = q[1], qz = q[2];
  int* o = out + ((size_t)b * S + s) * K;
  int cnt = 0, first = n - 1;
  for (int j = 0; j < n && cnt < K; ++j) {
    float dx = P[j * 3 + 0] - qx;
    float dy = P[j * 3 + 1] - qy;
    float dz = P[j * 3 + 2] - qz;
    float d2 = dx * dx + dy * dy + dz * dz;
    if (d2 <= r2) { if (cnt == 0) first = j; o[cnt++] = j; }
  }
  for (; cnt < K; ++cnt) o[cnt] = first;
}

// kNN via K lexicographic (dist, idx) min-scans; matches top_k tie order.
__global__ void knn_ker(const float* __restrict__ db, const float* __restrict__ q,
                        int n, int S, int K, int* __restrict__ oidx,
                        float* __restrict__ odist, long total) {
  long i = (long)blockIdx.x * blockDim.x + threadIdx.x;
  if (i >= total) return;
  int s = (int)(i % S);
  int b = (int)(i / S);
  const float* P = db + (size_t)b * n * 3;
  const float* Q = q + ((size_t)b * S + s) * 3;
  float qx = Q[0], qy = Q[1], qz = Q[2];
  int* oi = oidx + ((size_t)b * S + s) * K;
  float* od = odist ? odist + ((size_t)b * S + s) * K : (float*)0;
  float lastd = -1.0f; int lasti = -1;
  for (int kk = 0; kk < K; ++kk) {
    float bd = 1e30f; int bi = n - 1;
    for (int j = 0; j < n; ++j) {
      float dx = P[j * 3 + 0] - qx;
      float dy = P[j * 3 + 1] - qy;
      float dz = P[j * 3 + 2] - qz;
      float d2 = dx * dx + dy * dy + dz * dz;
      if (d2 > lastd || (d2 == lastd && j > lasti)) {
        if (d2 < bd) { bd = d2; bi = j; }
      }
    }
    oi[kk] = bi;
    if (od) od[kk] = bd;
    lastd = bd; lasti = bi;
  }
}

// grouped = concat(xyz[idx]-new_xyz, points[idx]) -> [S*K, 3+C]; per-batch ptrs.
__global__ void group_sa_ker(const float* __restrict__ xyz, const float* __restrict__ nxyz,
                             const float* __restrict__ pts, const int* __restrict__ idx,
                             int n, int S, int K, int C, float* __restrict__ out, long total) {
  long i = (long)blockIdx.x * blockDim.x + threadIdx.x;
  if (i >= total) return;
  int cw = 3 + C;
  int c = (int)(i % cw);
  long t = i / cw;
  int kk = (int)(t % K);
  int s  = (int)(t / K);
  int id = idx[(size_t)s * K + kk];
  if (id >= n) id = n - 1;
  float v;
  if (c < 3) v = xyz[(size_t)id * 3 + c] - nxyz[(size_t)s * 3 + c];
  else       v = pts[(size_t)id * C + (c - 3)];
  out[i] = v;
}

// flow-embedding group: concat(pos2[idx]-pos1, feat2[idx], feat1) -> [S*K, 3+2C]
__global__ void group_fe_ker(const float* __restrict__ pos2, const float* __restrict__ pos1,
                             const float* __restrict__ feat2, const float* __restrict__ feat1,
                             const int* __restrict__ idx, int n2, int S, int K, int C,
                             float* __restrict__ out, long total) {
  long i = (long)blockIdx.x * blockDim.x + threadIdx.x;
  if (i >= total) return;
  int cw = 3 + 2 * C;
  int c = (int)(i % cw);
  long t = i / cw;
  int kk = (int)(t % K);
  int s  = (int)(t / K);
  int id = idx[(size_t)s * K + kk];
  if (id >= n2) id = n2 - 1;
  float v;
  if (c < 3)          v = pos2[(size_t)id * 3 + c] - pos1[(size_t)s * 3 + c];
  else if (c < 3 + C) v = feat2[(size_t)id * C + (c - 3)];
  else                v = feat1[(size_t)s * C + (c - 3 - C)];
  out[i] = v;
}

// set_upconv group: concat(feat2[idx], pos2[idx]-pos1) -> [S*K, C2+3]
__global__ void group_su_ker(const float* __restrict__ pos2, const float* __restrict__ pos1,
                             const float* __restrict__ feat2, const int* __restrict__ idx,
                             int n2, int S, int K, int C2, float* __restrict__ out, long total) {
  long i = (long)blockIdx.x * blockDim.x + threadIdx.x;
  if (i >= total) return;
  int cw = C2 + 3;
  int c = (int)(i % cw);
  long t = i / cw;
  int kk = (int)(t % K);
  int s  = (int)(t / K);
  int id = idx[(size_t)s * K + kk];
  if (id >= n2) id = n2 - 1;
  float v;
  if (c < C2) v = feat2[(size_t)id * C2 + c];
  else        v = pos2[(size_t)id * 3 + (c - C2)] - pos1[(size_t)s * 3 + (c - C2)];
  out[i] = v;
}

// max over K: in [S*K, C] -> out [S, C]
__global__ void maxpool_ker(const float* __restrict__ X, int S, int K, int C,
                            float* __restrict__ out, long total) {
  long i = (long)blockIdx.x * blockDim.x + threadIdx.x;
  if (i >= total) return;
  int c = (int)(i % C);
  int s = (int)(i / C);
  const float* p = X + ((size_t)s * K) * C + c;
  float m = p[0];
  for (int k = 1; k < K; ++k) m = fmaxf(m, p[(size_t)k * C]);
  out[i] = m;
}

__global__ void concat_ker(const float* __restrict__ A, int Ca,
                           const float* __restrict__ Bp, int Cb, int S,
                           float* __restrict__ out, long total) {
  long i = (long)blockIdx.x * blockDim.x + threadIdx.x;
  if (i >= total) return;
  int cw = Ca + Cb;
  int c = (int)(i % cw);
  int s = (int)(i / cw);
  out[i] = (c < Ca) ? A[(size_t)s * Ca + c] : Bp[(size_t)s * Cb + (c - Ca)];
}

// 3-NN inverse-distance interpolation + concat feat1(3ch): out [S, C+3]
__global__ void fp_interp_ker(const float* __restrict__ feat2, const float* __restrict__ feat1,
                              const int* __restrict__ idx, const float* __restrict__ dist,
                              int n2, int S, int C, float* __restrict__ out, long total) {
  long i = (long)blockIdx.x * blockDim.x + threadIdx.x;
  if (i >= total) return;
  int cw = C + 3;
  int c = (int)(i % cw);
  int s = (int)(i / cw);
  if (c >= C) { out[i] = feat1[(size_t)s * 3 + (c - C)]; return; }
  const int* id = idx + (size_t)s * 3;
  const float* dd = dist + (size_t)s * 3;
  float w0 = 1.0f / fmaxf(dd[0], 1e-10f);
  float w1 = 1.0f / fmaxf(dd[1], 1e-10f);
  float w2 = 1.0f / fmaxf(dd[2], 1e-10f);
  float ws = w0 + w1 + w2;
  int i0 = id[0] < n2 ? id[0] : n2 - 1;
  int i1 = id[1] < n2 ? id[1] : n2 - 1;
  int i2 = id[2] < n2 ? id[2] : n2 - 1;
  float v = (w0 * feat2[(size_t)i0 * C + c] +
             w1 * feat2[(size_t)i1 * C + c] +
             w2 * feat2[(size_t)i2 * C + c]) / ws;
  out[i] = v;
}

// ---- WMMA fragment compute shared by main loop and tail ----
__device__ __forceinline__ v8f wmma_step(const _Float16* Ash, const _Float16* Bsh,
                                         int wave, int hi, int lr, v8f acc) {
  const v8h* Ap = (const v8h*)&Ash[(wave * 16 + lr) * 32 + hi * 8];
  v8h a0 = Ap[0];      // K = hi*8 .. +7
  v8h a1 = Ap[2];      // K = 16+hi*8 .. +7
  const v8h* Bq = (const v8h*)&Bsh[lr * 32 + hi * 8];
  v8h b0 = Bq[0];      // K = hi*8 .. +7
  v8h b1 = Bq[2];      // K = 16+hi*8 .. +7
  v16h a = __builtin_shufflevector(a0, a1, 0, 1, 2, 3, 4, 5, 6, 7,
                                   8, 9, 10, 11, 12, 13, 14, 15);
  v16h b = __builtin_shufflevector(b0, b1, 0, 1, 2, 3, 4, 5, 6, 7,
                                   8, 9, 10, 11, 12, 13, 14, 15);
  return __builtin_amdgcn_wmma_f32_16x16x32_f16(
      false, a, false, b, (short)0, acc, false, false);
}

// WMMA GEMM + bias + eval-BN + ReLU.  X [rows,cin] f32, W [cout,cin] f32.
// Block = 128 threads (4 waves); block tile 64(M) x 16(N); k-step 32.
// Main loop over cin&~31 uses precomputed pointers advanced by +32 floats
// (no clamps, no branches); single clamped tail step covers ragged cin.
__global__ __launch_bounds__(128) void gemm_bn_relu_ker(
    const float* __restrict__ X, int rows, int cin,
    const float* __restrict__ W, int cout,
    const float* __restrict__ gamma, const float* __restrict__ beta,
    const float* __restrict__ mean, const float* __restrict__ var,
    const float* __restrict__ bias, float* __restrict__ Y, int do_relu) {
  __shared__ __attribute__((aligned(16))) _Float16 Ash[64 * 32];
  __shared__ __attribute__((aligned(16))) _Float16 Bsh[16 * 32];
  int tid = threadIdx.x;
  int wave = tid >> 5;
  int lane = tid & 31;
  int hi = lane >> 4;
  int lr = lane & 15;
  int m0 = blockIdx.x * 64;
  int n0 = blockIdx.y * 16;

  // per-thread staging coordinates (row clamp folded into base pointers once)
  int kA = (tid * 4) & 31;                 // k offset within tile, 4-contig
  const float* aptr[4];
  int arow[4];
#pragma unroll
  for (int ch = 0; ch < 4; ++ch) {
    int idx = ch * 512 + tid * 4;
    int r = idx >> 5;
    arow[ch] = r;
    int gr = m0 + r;
    gr = gr < rows ? gr : rows - 1;        // rows % 64 == 0 in this network
    aptr[ch] = X + (size_t)gr * cin + kA;
  }
  int bn = (tid * 4) >> 5;
  int gn = n0 + bn;
  int gnc = gn < cout ? gn : cout - 1;
  bool bvalid = gn < cout;
  const float* bptr = W + (size_t)gnc * cin + kA;

  v8f acc = {};
  int cin_main = cin & ~31;

  for (int k0 = 0; k0 < cin_main; k0 += 32) {
#pragma unroll
    for (int ch = 0; ch < 4; ++ch) {       // A tile: 4 floats -> 1 ds_store_b64
      _Float16 h[4];
#pragma unroll
      for (int u = 0; u < 4; ++u) h[u] = (_Float16)aptr[ch][u];
      *(unsigned long long*)&Ash[arow[ch] * 32 + kA] = *(const unsigned long long*)h;
      aptr[ch] += 32;
    }
    {                                      // B tile (transposed [n][k])
      _Float16 h[4];
#pragma unroll
      for (int u = 0; u < 4; ++u) {
        float v = bptr[u];
        h[u] = (_Float16)(bvalid ? v : 0.0f);
      }
      *(unsigned long long*)&Bsh[bn * 32 + kA] = *(const unsigned long long*)h;
      bptr += 32;
    }
    __syncthreads();
    acc = wmma_step(Ash, Bsh, wave, hi, lr, acc);
    __syncthreads();
  }

  if (cin_main < cin) {                    // single ragged tail step
    int rem = cin - cin_main;              // 1..31 valid k values
#pragma unroll
    for (int ch = 0; ch < 4; ++ch) {
      _Float16 h[4];
#pragma unroll
      for (int u = 0; u < 4; ++u) {
        int kk = kA + u;
        int kc = kk < rem ? kk : rem - 1;  // clamp: load always valid
        float v = aptr[ch][kc - kA];
        h[u] = (_Float16)(kk < rem ? v : 0.0f);
      }
      *(unsigned long long*)&Ash[arow[ch] * 32 + kA] = *(const unsigned long long*)h;
    }
    {
      _Float16 h[4];
#pragma unroll
      for (int u = 0; u < 4; ++u) {
        int kk = kA + u;
        int kc = kk < rem ? kk : rem - 1;
        float v = bptr[kc - kA];
        h[u] = (_Float16)((kk < rem && bvalid) ? v : 0.0f);
      }
      *(unsigned long long*)&Bsh[bn * 32 + kA] = *(const unsigned long long*)h;
    }
    __syncthreads();
    acc = wmma_step(Ash, Bsh, wave, hi, lr, acc);
    __syncthreads();
  }

  int col = n0 + lr;
  if (col < cout) {
    float sc = gamma[col] * rsqrtf(var[col] + BN_EPS);
    float sh = beta[col] - mean[col] * sc;
    float bs = bias ? bias[col] : 0.0f;
#pragma unroll
    for (int r = 0; r < 8; ++r) {
      int row = m0 + wave * 16 + hi * 8 + r;
      if (row < rows) {
        float v = (acc[r] + bs) * sc + sh;
        if (do_relu) v = fmaxf(v, 0.0f);
        Y[(size_t)row * cout + col] = v;
      }
    }
  }
}

// final 128->3 conv + transpose to [B,3,N]
__global__ void conv2_ker(const float* __restrict__ h, const float* __restrict__ W,
                          const float* __restrict__ bvec, float* __restrict__ out,
                          int N, long total) {
  long i = (long)blockIdx.x * blockDim.x + threadIdx.x;
  if (i >= total) return;
  int nn = (int)(i % N);
  int b  = (int)(i / N);
  const float* x = h + ((size_t)b * N + nn) * 128;
  for (int o = 0; o < 3; ++o) {
    float s = bvec[o];
    const float* w = W + o * 128;
    for (int c = 0; c < 128; ++c) s += x[c] * w[c];
    out[((size_t)b * 3 + o) * N + nn] = s;
  }
}

// ------------------------------ host helpers -------------------------------

struct Layer {
  const float *w, *beta, *gamma, *mean, *var, *bias;
  int cin, cout;
};

static void read_layer(Layer* L, void* const* din, int* i, int cin, int cout, bool bias) {
  L->bias  = bias ? (const float*)din[(*i)++] : (const float*)0;
  L->beta  = (const float*)din[(*i)++];
  L->gamma = (const float*)din[(*i)++];
  L->mean  = (const float*)din[(*i)++];
  L->var   = (const float*)din[(*i)++];
  L->w     = (const float*)din[(*i)++];
  L->cin = cin; L->cout = cout;
}

static inline unsigned gsz(long total) { return (unsigned)((total + 255) / 256); }

static const float* run_mlp(const Layer* L, int nl, const float* X, int rows,
                            float* sa, float* sb, float* dst, hipStream_t stream) {
  const float* cur = X;
  for (int l = 0; l < nl; ++l) {
    float* nxt;
    if (dst && l == nl - 1) nxt = dst;
    else nxt = (cur == sa) ? sb : sa;
    dim3 g((rows + 63) / 64, (L[l].cout + 15) / 16);
    gemm_bn_relu_ker<<<g, 128, 0, stream>>>(cur, rows, L[l].cin, L[l].w, L[l].cout,
                                            L[l].gamma, L[l].beta, L[l].mean, L[l].var,
                                            L[l].bias, nxt, 1);
    cur = nxt;
  }
  return cur;
}

struct Ctx {
  hipStream_t stream;
  int* fpsidx;      // [B,1024]
  float* fpsdist;   // [B,4096]
  int* idxbuf;      // [B*1024*16] max
  float* distbuf;   // [B*4096*3]
  float* SA;        // scratch ping
  float* SB;        // scratch pong
  float* TP;        // pooled temp
  float* TQ;        // concat temp
};

static void do_sa(Ctx* c, const float* xyz, int n, const float* pts, int Cpts,
                  int S, int K, float radius, const Layer* L, int nl,
                  float* nxyz, float* nfeat) {
  hipStream_t stream = c->stream;
  fps_ker<<<NB, 256, 0, stream>>>(xyz, n, S, c->fpsidx, c->fpsdist);
  long tg = (long)NB * S * 3;
  gather_xyz_ker<<<gsz(tg), 256, 0, stream>>>(xyz, c->fpsidx, n, S, nxyz, tg);
  long tq = (long)NB * S;
  ballq_ker<<<gsz(tq), 256, 0, stream>>>(xyz, nxyz, n, S, K, radius * radius,
                                         c->idxbuf, tq);
  int cw = 3 + Cpts;
  int Cout = L[nl - 1].cout;
  for (int b = 0; b < NB; ++b) {
    long tg2 = (long)S * K * cw;
    group_sa_ker<<<gsz(tg2), 256, 0, stream>>>(
        xyz + (size_t)b * n * 3, nxyz + (size_t)b * S * 3,
        pts + (size_t)b * n * Cpts, c->idxbuf + (size_t)b * S * K,
        n, S, K, Cpts, c->SA, tg2);
    const float* y = run_mlp(L, nl, c->SA, S * K, c->SA, c->SB, 0, stream);
    long tm = (long)S * Cout;
    maxpool_ker<<<gsz(tm), 256, 0, stream>>>(y, S, K, Cout,
                                             nfeat + (size_t)b * S * Cout, tm);
  }
}

static void do_su(Ctx* c, const float* pos1, int S, const float* pos2, int n2,
                  const float* feat1, int C1, const float* feat2, int C2, int K,
                  const Layer* L1, int nl1, const Layer* L2, int nl2, float* out) {
  hipStream_t stream = c->stream;
  long tq = (long)NB * S;
  knn_ker<<<gsz(tq), 256, 0, stream>>>(pos2, pos1, n2, S, K, c->idxbuf, (float*)0, tq);
  int cw = C2 + 3;
  int Cout2 = L2[nl2 - 1].cout;
  for (int b = 0; b < NB; ++b) {
    long tg = (long)S * K * cw;
    group_su_ker<<<gsz(tg), 256, 0, stream>>>(
        pos2 + (size_t)b * n2 * 3, pos1 + (size_t)b * S * 3,
        feat2 + (size_t)b * n2 * C2, c->idxbuf + (size_t)b * S * K,
        n2, S, K, C2, c->SA, tg);
    const float* y;
    int Cm;
    if (nl1 > 0) {
      y = run_mlp(L1, nl1, c->SA, S * K, c->SA, c->SB, 0, stream);
      Cm = L1[nl1 - 1].cout;
    } else {
      y = c->SA; Cm = cw;
    }
    long tm = (long)S * Cm;
    maxpool_ker<<<gsz(tm), 256, 0, stream>>>(y, S, K, Cm, c->TP, tm);
    long tc = (long)S * (Cm + C1);
    concat_ker<<<gsz(tc), 256, 0, stream>>>(c->TP, Cm, feat1 + (size_t)b * S * C1,
                                            C1, S, c->TQ, tc);
    run_mlp(L2, nl2, c->TQ, S, c->SA, c->SB, out + (size_t)b * S * Cout2, stream);
  }
}

// ------------------------------ entry point --------------------------------

extern "C" void kernel_launch(void* const* d_in, const int* in_sizes, int n_in,
                              void* d_out, int out_size, void* d_ws, size_t ws_size,
                              hipStream_t stream) {
  (void)in_sizes; (void)n_in; (void)out_size; (void)ws_size;

  const float* pc1 = (const float*)d_in[0];
  const float* pc2 = (const float*)d_in[1];
  const float* fe1 = (const float*)d_in[2];
  const float* fe2 = (const float*)d_in[3];

  // params flattened in sorted-key pytree order
  int ii = 4;
  const float* conv2_b = (const float*)d_in[ii++];
  const float* conv2_w = (const float*)d_in[ii++];
  Layer Lfe[3];  read_layer(&Lfe[0], d_in, &ii, 259, 128, false);
                 read_layer(&Lfe[1], d_in, &ii, 128, 128, false);
                 read_layer(&Lfe[2], d_in, &ii, 128, 128, false);
  Layer Lfp[2];  read_layer(&Lfp[0], d_in, &ii, 259, 256, true);
                 read_layer(&Lfp[1], d_in, &ii, 256, 256, true);
  Layer Lhead[1];read_layer(&Lhead[0], d_in, &ii, 256, 128, false);
  Layer Lsa1[3]; read_layer(&Lsa1[0], d_in, &ii,   6,  32, false);
                 read_layer(&Lsa1[1], d_in, &ii,  32,  32, false);
                 read_layer(&Lsa1[2], d_in, &ii,  32,  64, false);
  Layer Lsa2[3]; read_layer(&Lsa2[0], d_in, &ii,  67,  64, false);
                 read_layer(&Lsa2[1], d_in, &ii,  64,  64, false);
                 read_layer(&Lsa2[2], d_in, &ii,  64, 128, false);
  Layer Lsa3[3]; read_layer(&Lsa3[0], d_in, &ii, 131, 128, false);
                 read_layer(&Lsa3[1], d_in, &ii, 128, 128, false);
                 read_layer(&Lsa3[2], d_in, &ii, 128, 256, false);
  Layer Lsa4[3]; read_layer(&Lsa4[0], d_in, &ii, 259, 256, false);
                 read_layer(&Lsa4[1], d_in, &ii, 256, 256, false);
                 read_layer(&Lsa4[2], d_in, &ii, 256, 512, false);
  // su1_mlp1 is empty
  Layer Lsu1b[2];read_layer(&Lsu1b[0], d_in, &ii, 771, 256, false);
                 read_layer(&Lsu1b[1], d_in, &ii, 256, 256, false);
  Layer Lsu2a[3];read_layer(&Lsu2a[0], d_in, &ii, 259, 128, false);
                 read_layer(&Lsu2a[1], d_in, &ii, 128, 128, false);
                 read_layer(&Lsu2a[2], d_in, &ii, 128, 256, false);
  Layer Lsu2b[1];read_layer(&Lsu2b[0], d_in, &ii, 512, 256, false);
  Layer Lsu3a[3];read_layer(&Lsu3a[0], d_in, &ii, 259, 128, false);
                 read_layer(&Lsu3a[1], d_in, &ii, 128, 128, false);
                 read_layer(&Lsu3a[2], d_in, &ii, 128, 256, false);
  Layer Lsu3b[1];read_layer(&Lsu3b[0], d_in, &ii, 320, 256, false);

  // ----- workspace carve-up (~115 MB total) -----
  char* wsp = (char*)d_ws;
  auto alloc_f = [&](size_t n) -> float* {
    float* p = (float*)wsp; wsp += ((n * 4 + 255) & ~(size_t)255); return p;
  };
  auto alloc_i = [&](size_t n) -> int* {
    int* p = (int*)wsp; wsp += ((n * 4 + 255) & ~(size_t)255); return p;
  };

  float* x1 = alloc_f((size_t)NB * NP * 3);
  float* x2 = alloc_f((size_t)NB * NP * 3);
  float* f1 = alloc_f((size_t)NB * NP * 3);
  float* f2 = alloc_f((size_t)NB * NP * 3);
  float* l1x1 = alloc_f((size_t)NB * 1024 * 3);
  float* l1x2 = alloc_f((size_t)NB * 1024 * 3);
  float* l1f1 = alloc_f((size_t)NB * 1024 * 64);
  float* l1f2 = alloc_f((size_t)NB * 1024 * 64);
  float* l2x1 = alloc_f((size_t)NB * 256 * 3);
  float* l2x2 = alloc_f((size_t)NB * 256 * 3);
  float* l2f1 = alloc_f((size_t)NB * 256 * 128);
  float* l2f2 = alloc_f((size_t)NB * 256 * 128);
  float* l2fnew = alloc_f((size_t)NB * 256 * 128);
  float* l3x = alloc_f((size_t)NB * 64 * 3);
  float* l3f = alloc_f((size_t)NB * 64 * 256);
  float* l4x = alloc_f((size_t)NB * 16 * 3);
  float* l4f = alloc_f((size_t)NB * 16 * 512);
  float* l3fn = alloc_f((size_t)NB * 64 * 256);
  float* l2cat = alloc_f((size_t)NB * 256 * 256);
  float* l2fn = alloc_f((size_t)NB * 256 * 256);
  float* l1fn = alloc_f((size_t)NB * 1024 * 256);
  float* l0fn = alloc_f((size_t)NB * NP * 256);
  float* hbuf = alloc_f((size_t)NB * NP * 128);

  Ctx c;
  c.stream = stream;
  c.fpsidx = alloc_i((size_t)NB * 1024);
  c.fpsdist = alloc_f((size_t)NB * NP);
  c.idxbuf = alloc_i((size_t)NB * 1024 * 16);
  c.distbuf = alloc_f((size_t)NB * NP * 3);
  c.TP = alloc_f((size_t)1024 * 256);
  c.TQ = alloc_f((size_t)1024 * 320);
  c.SA = alloc_f((size_t)16384 * 288);   // >= 16384*259
  c.SB = alloc_f((size_t)16384 * 288);

  // ----- transpose inputs to channels-last -----
  long tt = (long)NB * NP * 3;
  chlast_ker<<<gsz(tt), 256, 0, stream>>>(pc1, x1, 3, NP, tt);
  chlast_ker<<<gsz(tt), 256, 0, stream>>>(pc2, x2, 3, NP, tt);
  chlast_ker<<<gsz(tt), 256, 0, stream>>>(fe1, f1, 3, NP, tt);
  chlast_ker<<<gsz(tt), 256, 0, stream>>>(fe2, f2, 3, NP, tt);

  // ----- set abstraction pyramid, cloud 1 and cloud 2 -----
  do_sa(&c, x1, NP, f1, 3, 1024, 16, 0.004f, Lsa1, 3, l1x1, l1f1);
  do_sa(&c, l1x1, 1024, l1f1, 64, 256, 16, 0.008f, Lsa2, 3, l2x1, l2f1);
  do_sa(&c, x2, NP, f2, 3, 1024, 16, 0.004f, Lsa1, 3, l1x2, l1f2);
  do_sa(&c, l1x2, 1024, l1f2, 64, 256, 16, 0.008f, Lsa2, 3, l2x2, l2f2);

  // ----- flow embedding (k=64 over cloud2 level-2) -----
  {
    long tq = (long)NB * 256;
    knn_ker<<<gsz(tq), 256, 0, stream>>>(l2x2, l2x1, 256, 256, 64,
                                         c.idxbuf, (float*)0, tq);
    for (int b = 0; b < NB; ++b) {
      long tg = (long)256 * 64 * 259;
      group_fe_ker<<<gsz(tg), 256, 0, stream>>>(
          l2x2 + (size_t)b * 256 * 3, l2x1 + (size_t)b * 256 * 3,
          l2f2 + (size_t)b * 256 * 128, l2f1 + (size_t)b * 256 * 128,
          c.idxbuf + (size_t)b * 256 * 64, 256, 256, 64, 128, c.SA, tg);
      const float* y = run_mlp(Lfe, 3, c.SA, 256 * 64, c.SA, c.SB, 0, stream);
      long tm = (long)256 * 128;
      maxpool_ker<<<gsz(tm), 256, 0, stream>>>(y, 256, 64, 128,
                                               l2fnew + (size_t)b * 256 * 128, tm);
    }
  }

  // ----- deeper abstraction on embedded features -----
  do_sa(&c, l2x1, 256, l2fnew, 128, 64, 8, 0.016f, Lsa3, 3, l3x, l3f);
  do_sa(&c, l3x, 64, l3f, 256, 16, 8, 0.032f, Lsa4, 3, l4x, l4f);

  // ----- set upconv chain -----
  do_su(&c, l3x, 64, l4x, 16, l3f, 256, l4f, 512, 8, (Layer*)0, 0, Lsu1b, 2, l3fn);
  {
    long tc = (long)NB * 256 * 256;
    concat_ker<<<gsz(tc), 256, 0, stream>>>(l2f1, 128, l2fnew, 128, NB * 256,
                                            l2cat, tc);
  }
  do_su(&c, l2x1, 256, l3x, 64, l2cat, 256, l3fn, 256, 8, Lsu2a, 3, Lsu2b, 1, l2fn);
  do_su(&c, l1x1, 1024, l2x1, 256, l1f1, 64, l2fn, 256, 8, Lsu3a, 3, Lsu3b, 1, l1fn);

  // ----- feature propagation (3-NN interpolation to full resolution) -----
  {
    long tq = (long)NB * NP;
    knn_ker<<<gsz(tq), 256, 0, stream>>>(l1x1, x1, 1024, NP, 3,
                                         c.idxbuf, c.distbuf, tq);
    for (int b = 0; b < NB; ++b) {
      long ti = (long)NP * 259;
      fp_interp_ker<<<gsz(ti), 256, 0, stream>>>(
          l1fn + (size_t)b * 1024 * 256, f1 + (size_t)b * NP * 3,
          c.idxbuf + (size_t)b * NP * 3, c.distbuf + (size_t)b * NP * 3,
          1024, NP, 256, c.SA, ti);
      run_mlp(Lfp, 2, c.SA, NP, c.SA, c.SB, l0fn + (size_t)b * NP * 256, stream);
    }
  }

  // ----- head: conv1+bn+relu (full batch, one WMMA GEMM), then 128->3 conv -----
  run_mlp(Lhead, 1, l0fn, NB * NP, c.SA, c.SB, hbuf, stream);
  long to = (long)NB * NP;
  conv2_ker<<<gsz(to), 256, 0, stream>>>(hbuf, conv2_w, conv2_b, (float*)d_out, NP, to);
}
